// CustomLSTMLayer_11355893531349
// MI455X (gfx1250) — matrix operand order repaired and checked by reading, not compiled, not run on hardware
//
#include <hip/hip_runtime.h>

// ---------------------------------------------------------------------------
// 2-layer LSTM, MI455X (gfx1250, wave32, WMMA), single fused persistent grid.
// B=64, T=512, D_IN=128, H=256.
//
// Partition: 16 WGs total = 2 layers x 8 hidden-slices. WG (layer, w) owns
// hidden units j in [32w, 32w+32) -> gate columns {g*256 + [32w,32w+32)} for
// all 64 batch rows. Its bf16 weight slice (Kc x 128) lives in LDS for the
// entire recurrence; per-step operands are pure LDS -> v_wmma traffic.
// h is exchanged through L2 ping-pong buffers with a 16-WG split barrier;
// layer 1 runs one step behind layer 0 (software pipeline), so the whole
// network is 513 barrier steps in one launch.
// ---------------------------------------------------------------------------

typedef __attribute__((ext_vector_type(16))) __bf16 v16bf;
typedef __attribute__((ext_vector_type(4)))  __bf16 v4bf;
typedef __attribute__((ext_vector_type(8)))  float  v8f;
typedef __attribute__((ext_vector_type(4)))  float  v4f;

#define B_    64
#define T_    512
#define DIN_  128
#define H_    256
#define NWG_  16
#define KC0_  (H_ + DIN_)      // 384: layer0 A = [h | x]
#define KC1_  (2 * H_)         // 512: layer1 A = [h | h0]
#define WLDS_BYTES 131072      // LDS weight region (max: 512*128*2)
#define SMEM_BYTES (WLDS_BYTES + 64 * KC1_ * 2)   // + A region = 192 KB

__device__ __forceinline__ float sigmoidf_(float v) {
  return 1.0f / (1.0f + __expf(-v));
}

// ---------------------------------------------------------------------------
// Per-launch init: zero the h ping-pong buffers and the barrier state.
// ---------------------------------------------------------------------------
__global__ void lstm_init_kernel(int* __restrict__ bar,
                                 __bf16* __restrict__ hex) {
  int idx = blockIdx.x * blockDim.x + threadIdx.x;
  if (idx < 4 * B_ * H_) hex[idx] = (__bf16)0.0f;   // hex[layer][parity][b][j]
  if (idx == 0) { bar[0] = 0; bar[1] = 0; }
}

// ---------------------------------------------------------------------------
// The recurrence body, specialized per layer (KC compile-time).
// ---------------------------------------------------------------------------
template <int KC, int LAYER>
__device__ void lstm_run(const float* __restrict__ x,
                         const float* __restrict__ W,     // (1024, KC) fp32
                         const float* __restrict__ bias,  // (1024,)
                         __bf16* __restrict__ hex,        // 4 * 16384 bf16
                         int* __restrict__ bar,
                         float* __restrict__ out,
                         char* smem) {
  __bf16* Wlds = (__bf16*)smem;                  // KC x 128 (tile-packed)
  __bf16* Alds = (__bf16*)(smem + WLDS_BYTES);   // 64 x KC   ([h | x])

  const int tid   = threadIdx.x;
  const int wave  = tid >> 5;
  const int lane  = tid & 31;
  const int kHalf = (lane >> 4) & 1;
  const int nloc  = lane & 15;
  const int wg    = blockIdx.x & 7;      // hidden slice [32*wg, 32*wg+32)
  const int mtile = wave >> 1;           // batch tile: rows [16*mtile, +16)
  const int sTile = wave & 1;            // which 16 of the 32 hidden cols
  const int jcol  = wg * 32 + sTile * 16 + nloc;   // this lane's hidden unit

  // Per-lane bias for the 4 gate chunks (f, i, o, g) at column jcol.
  float bv[4];
#pragma unroll
  for (int g = 0; g < 4; ++g) bv[g] = bias[g * H_ + jcol];

  // One-time: transpose + cast this WG's fp32 weight slice into LDS.
  // LDS layout: Wlds[k][tile*16 + c], tile = g*2+s -> global col
  // n = g*256 + 32*wg + 16*s + c.  (W is row-major (1024, KC): W[n][k].)
  for (int i = tid; i < KC * 128; i += 256) {
    int k = i >> 7, cloc = i & 127;
    int tile = cloc >> 4, c = cloc & 15;
    int n = (tile >> 1) * H_ + wg * 32 + (tile & 1) * 16 + c;
    Wlds[i] = (__bf16)W[(size_t)n * KC + k];
  }

  float creg[8];   // fp32 cell state, WMMA C/D register layout
#pragma unroll
  for (int r = 0; r < 8; ++r) creg[r] = 0.0f;

  const __bf16* hexSelf = hex + LAYER * 2 * (B_ * H_);
  __bf16*       hexSelfW = hex + LAYER * 2 * (B_ * H_);
  const __bf16* hex0 = hex;   // layer0 exchange = layer1's x input

  for (int s = 0; s <= T_; ++s) {
    const int  t      = (LAYER == 0) ? s : s - 1;
    const bool active = (LAYER == 0) ? (s < T_) : (s >= 1);

    if (active) {
      // ---- fill A h-region: full 64x256 h_{t-1} of own layer (32 KB) ----
      {
        const __bf16* hsrc = hexSelf + ((t - 1) & 1) * (B_ * H_);
        for (int i = tid; i < B_ * 32; i += 256) {      // 64 rows x 32 x 16B
          int b = i >> 5, q = i & 31;
          *(uint4*)((char*)Alds + (size_t)b * (KC * 2) + q * 16) =
              *(const uint4*)((const char*)hsrc + (size_t)b * 512 + q * 16);
        }
      }
      // ---- fill A x-region ----
      if (LAYER == 0) {
        // x[:, t, :] fp32 -> bf16  (64 x 128)
        for (int i = tid; i < B_ * 32; i += 256) {      // 64 rows x 32 float4
          int b = i >> 5, q = i & 31;
          v4f v = *(const v4f*)(x + (size_t)b * (T_ * DIN_) +
                                (size_t)t * DIN_ + q * 4);
          v4bf hb;
          hb[0] = (__bf16)v[0]; hb[1] = (__bf16)v[1];
          hb[2] = (__bf16)v[2]; hb[3] = (__bf16)v[3];
          *(v4bf*)&Alds[b * KC + H_ + q * 4] = hb;
        }
        if (t + 1 < T_)
          __builtin_prefetch(x + (size_t)(tid >> 2) * (T_ * DIN_) +
                                 (size_t)(t + 1) * DIN_ + (tid & 3) * 32, 0, 1);
      } else {
        // h0_t bf16 (written by layer0 last iteration)  (64 x 256)
        const __bf16* xsrc = hex0 + (t & 1) * (B_ * H_);
        for (int i = tid; i < B_ * 32; i += 256) {
          int b = i >> 5, q = i & 31;
          *(uint4*)((char*)Alds + (size_t)b * (KC * 2) + H_ * 2 + q * 16) =
              *(const uint4*)((const char*)xsrc + (size_t)b * 512 + q * 16);
        }
      }
      __syncthreads();   // A (and, on iter 0, Wlds) ready

      // ---- gates = A @ W : pure LDS -> WMMA ----
      v8f acc[4];
      const v8f vz = {0.f, 0.f, 0.f, 0.f, 0.f, 0.f, 0.f, 0.f};
      acc[0] = vz; acc[1] = vz; acc[2] = vz; acc[3] = vz;
#pragma unroll 4
      for (int kb = 0; kb < (KC >> 5); ++kb) {
        v16bf a = *(const v16bf*)
            &Alds[(mtile * 16 + nloc) * KC + kb * 32 + kHalf * 16];
#pragma unroll
        for (int g = 0; g < 4; ++g) {
          v16bf bf;
#pragma unroll
          for (int e = 0; e < 16; ++e)
            bf[e] = Wlds[(kb * 32 + kHalf * 16 + e) * 128 + (g * 2 + sTile) * 16 + nloc];
          acc[g] = __builtin_amdgcn_wmma_f32_16x16x32_bf16(
              false, a, false, bf, (short)0, acc[g], false, false);
        }
      }

      // ---- elementwise LSTM update; publish h slice to L2 ping-pong ----
      __bf16* hdst = hexSelfW + (t & 1) * (B_ * H_);
#pragma unroll
      for (int r = 0; r < 8; ++r) {
        const int m = mtile * 16 + kHalf * 8 + r;        // batch row
        float f  = sigmoidf_(acc[0][r] + bv[0]);
        float ig = sigmoidf_(acc[1][r] + bv[1]);
        float o  = sigmoidf_(acc[2][r] + bv[2]);
        float gg = tanhf    (acc[3][r] + bv[3]);
        float cn = f * creg[r] + ig * gg;
        float hn = o * tanhf(cn);
        creg[r] = cn;
        hdst[m * H_ + jcol] = (__bf16)hn;
        if (t == T_ - 1) {
          if (LAYER == 0) {
            out[16384 + m * H_ + jcol] = hn;   // h_n[0]
            out[49152 + m * H_ + jcol] = cn;   // c_n[0]
          } else {
            out[        m * H_ + jcol] = hn;   // h1T (primary output)
            out[32768 + m * H_ + jcol] = hn;   // h_n[1]
            out[65536 + m * H_ + jcol] = cn;   // c_n[1]
          }
        }
      }
    }

    // ---- 16-WG device split-barrier (monotonic phase, re-init per launch) --
    __threadfence();
    __syncthreads();
    if (tid == 0) {
      int old = __hip_atomic_fetch_add(&bar[0], 1, __ATOMIC_ACQ_REL,
                                       __HIP_MEMORY_SCOPE_AGENT);
      if (old == NWG_ * (s + 1) - 1) {
        __hip_atomic_store(&bar[1], s + 1, __ATOMIC_RELEASE,
                           __HIP_MEMORY_SCOPE_AGENT);
      } else {
        while (__hip_atomic_load(&bar[1], __ATOMIC_ACQUIRE,
                                 __HIP_MEMORY_SCOPE_AGENT) < s + 1)
          __builtin_amdgcn_s_sleep(1);
      }
    }
    __syncthreads();
    __threadfence();
  }
}

__launch_bounds__(256, 1)
__global__ void lstm_fused_kernel(const float* __restrict__ x,
                                  const float* __restrict__ W0,
                                  const float* __restrict__ b0,
                                  const float* __restrict__ W1,
                                  const float* __restrict__ b1,
                                  __bf16* __restrict__ hex,
                                  int* __restrict__ bar,
                                  float* __restrict__ out) {
  extern __shared__ char smem[];
  if ((blockIdx.x >> 3) == 0)
    lstm_run<KC0_, 0>(x, W0, b0, hex, bar, out, smem);
  else
    lstm_run<KC1_, 1>(x, W1, b1, hex, bar, out, smem);
}

// ---------------------------------------------------------------------------
extern "C" void kernel_launch(void* const* d_in, const int* in_sizes, int n_in,
                              void* d_out, int out_size, void* d_ws,
                              size_t ws_size, hipStream_t stream) {
  const float* x  = (const float*)d_in[0];  // (B,T,DIN)
  const float* W0 = (const float*)d_in[1];  // (1024,384)
  const float* b0 = (const float*)d_in[2];  // (1024,)
  const float* W1 = (const float*)d_in[3];  // (1024,512)
  const float* b1 = (const float*)d_in[4];  // (1024,)
  float* out = (float*)d_out;
  (void)in_sizes; (void)n_in; (void)out_size; (void)ws_size;

  // Workspace: [barrier: 2 ints (16B pad)] [hex: 4 x 64 x 256 bf16 = 128 KB]
  int*    bar = (int*)d_ws;
  __bf16* hex = (__bf16*)((char*)d_ws + 16);

  static_assert(SMEM_BYTES == 196608, "LDS budget");
  (void)hipFuncSetAttribute((const void*)lstm_fused_kernel,
                            hipFuncAttributeMaxDynamicSharedMemorySize,
                            SMEM_BYTES);

  lstm_init_kernel<<<(4 * B_ * H_ + 255) / 256, 256, 0, stream>>>(bar, hex);
  lstm_fused_kernel<<<NWG_, 256, SMEM_BYTES, stream>>>(x, W0, b0, W1, b1,
                                                       hex, bar, out);
}
